// GraphBlock_45449343926481
// MI455X (gfx1250) — matrix operand order repaired and checked by reading, not compiled, not run on hardware
//
#include <hip/hip_runtime.h>
#include <hip/hip_bf16.h>

typedef __attribute__((ext_vector_type(16))) _Float16 v16h;
typedef __attribute__((ext_vector_type(8)))  float    v8f;

#define HH  64     // H * HID
#define DN  128    // D_NODE == D_EDGE
#define DO  128    // D_OUT
#define LN_EPS 1e-5f

__device__ __forceinline__ float gelu_erf(float x) {
    return 0.5f * x * (1.0f + erff(x * 0.7071067811865475f));
}

// ---------------------------------------------------------------------
// A fragment (16x32 f16): lane<16 -> row=lane,    K = k0+{0..7,16..23}
//                         lane>=16 -> row=lane-16, K = k0+{8..15,24..31}
// Unconditional: caller clamps the row; garbage rows are never stored.
// Four b128 loads + pack-converts.
// ---------------------------------------------------------------------
__device__ __forceinline__ v16h load_a_frag(const float* __restrict__ X, long row,
                                            int ldx, int k0, int lane) {
    const float* p = X + row * (long)ldx + k0 + 8 * ((lane >> 4) & 1);
    const float4 a0 = *(const float4*)(p + 0);
    const float4 a1 = *(const float4*)(p + 4);
    const float4 b0 = *(const float4*)(p + 16);
    const float4 b1 = *(const float4*)(p + 20);
    v16h a;
    a[0] = (_Float16)a0.x; a[1] = (_Float16)a0.y; a[2] = (_Float16)a0.z; a[3] = (_Float16)a0.w;
    a[4] = (_Float16)a1.x; a[5] = (_Float16)a1.y; a[6] = (_Float16)a1.z; a[7] = (_Float16)a1.w;
    a[8]  = (_Float16)b0.x; a[9]  = (_Float16)b0.y; a[10] = (_Float16)b0.z; a[11] = (_Float16)b0.w;
    a[12] = (_Float16)b1.x; a[13] = (_Float16)b1.y; a[14] = (_Float16)b1.z; a[15] = (_Float16)b1.w;
    return a;
}

// ---------------------------------------------------------------------
// Pre-pack a weight matrix W[K][N] (f32 row-major) into f16 fragments in
// exact WMMA B-register order:  out[((kk*NT + t)*32 + lane)*16 + e]
//   col = t*16 + (lane&15);  k = kk*32 + 8*(lane>>4) + (e&7) + 16*(e>>3)
// GEMM kernels then fetch one contiguous 32-byte v16h per (kk,t,lane).
// ---------------------------------------------------------------------
__global__ __launch_bounds__(256) void pack_w_kernel(
    const float* __restrict__ W, int K, int N, _Float16* __restrict__ out)
{
    const int idx = blockIdx.x * blockDim.x + threadIdx.x;
    if (idx >= K * N) return;
    const int NT   = N >> 4;
    const int e    = idx & 15;
    const int lane = (idx >> 4) & 31;
    const int rest = idx >> 9;               // kk*NT + t
    const int t    = rest % NT;
    const int kk   = rest / NT;
    const int col  = t * 16 + (lane & 15);
    const int k    = kk * 32 + ((lane >> 4) & 1) * 8 + (e & 7) + (e >> 3) * 16;
    out[idx] = (_Float16)W[(long)k * N + col];
}

// =====================================================================
// Kernel 1: hv = LN(GELU(node_feats @ W_node))   [nrows x 64]
// 256 thr (8 waves), 128 rows / block, 16-row stripe per wave.
// =====================================================================
__global__ __launch_bounds__(256) void node_proj_kernel(
    const float* __restrict__ X, const v16h* __restrict__ Wp,
    const float* __restrict__ g, const float* __restrict__ b,
    float* __restrict__ hv, int nrows)
{
    __shared__ float tile[128 * 65];
    const int wave = threadIdx.x >> 5;
    const int lane = threadIdx.x & 31;
    const int h = lane >> 4, m = lane & 15;
    const int blk0 = blockIdx.x * 128;
    long row = blk0 + wave * 16 + m;
    if (row >= nrows) row = nrows - 1;        // clamp; result discarded

    v8f z = {};
    v8f acc[4];
#pragma unroll
    for (int t = 0; t < 4; ++t) acc[t] = z;

#pragma unroll
    for (int kk = 0; kk < 4; ++kk) {
        const v16h a = load_a_frag(X, row, DN, kk * 32, lane);
#pragma unroll
        for (int t = 0; t < 4; ++t) {
            const v16h bf = Wp[(kk * 4 + t) * 32 + lane];   // NT=4
            acc[t] = __builtin_amdgcn_wmma_f32_16x16x32_f16(
                false, a, false, bf, (short)0, acc[t], false, false);
        }
    }

#pragma unroll
    for (int t = 0; t < 4; ++t)
#pragma unroll
        for (int r = 0; r < 8; ++r)
            tile[(wave * 16 + 8 * h + r) * 65 + t * 16 + m] = acc[t][r];
    __syncthreads();

    if (threadIdx.x < 128) {
        const long grow = blk0 + threadIdx.x;
        if (grow < nrows) {
            float* rp = &tile[threadIdx.x * 65];
            float s = 0.f;
#pragma unroll
            for (int c = 0; c < HH; ++c) { float x = gelu_erf(rp[c]); rp[c] = x; s += x; }
            const float mu = s * (1.f / HH);
            float vs = 0.f;
#pragma unroll
            for (int c = 0; c < HH; ++c) { const float d = rp[c] - mu; vs += d * d; }
            const float inv = rsqrtf(vs * (1.f / HH) + LN_EPS);
            float* op = hv + grow * HH;
#pragma unroll
            for (int c = 0; c < HH; c += 4) {
                float4 o;
                o.x = (rp[c + 0] - mu) * inv * g[c + 0] + b[c + 0];
                o.y = (rp[c + 1] - mu) * inv * g[c + 1] + b[c + 1];
                o.z = (rp[c + 2] - mu) * inv * g[c + 2] + b[c + 2];
                o.w = (rp[c + 3] - mu) * inv * g[c + 3] + b[c + 3];
                *(float4*)(op + c) = o;
            }
        }
    }
}

// =====================================================================
// Kernel 2 (fused): he = exp(LN(edge_feats @ W_edge)) per 128-edge tile,
// then m = hv[src]*he scattered via atomicAdd into hsum[dst].
// `he` never touches HBM; hv & hsum stay L2-resident (12.8 MB each).
// =====================================================================
__global__ __launch_bounds__(256) void edge_fused_kernel(
    const float* __restrict__ Xe, const v16h* __restrict__ Wp,
    const float* __restrict__ g, const float* __restrict__ b,
    const int* __restrict__ src, const int* __restrict__ dst,
    const float* __restrict__ hv, float* __restrict__ hsum, int nedges)
{
    __shared__ float tile[128 * 65];
    const int wave = threadIdx.x >> 5;
    const int lane = threadIdx.x & 31;
    const int h = lane >> 4, m = lane & 15;
    const int blk0 = blockIdx.x * 128;
    long row = blk0 + wave * 16 + m;
    if (row >= nedges) row = nedges - 1;      // clamp; result discarded

    // warm caches for the next 128-edge tile (global_prefetch_b8)
    __builtin_prefetch(Xe + (row + 128) * (long)DN, 0, 1);

    v8f z = {};
    v8f acc[4];
#pragma unroll
    for (int t = 0; t < 4; ++t) acc[t] = z;

#pragma unroll
    for (int kk = 0; kk < 4; ++kk) {
        const v16h a = load_a_frag(Xe, row, DN, kk * 32, lane);
#pragma unroll
        for (int t = 0; t < 4; ++t) {
            const v16h bf = Wp[(kk * 4 + t) * 32 + lane];   // NT=4
            acc[t] = __builtin_amdgcn_wmma_f32_16x16x32_f16(
                false, a, false, bf, (short)0, acc[t], false, false);
        }
    }

#pragma unroll
    for (int t = 0; t < 4; ++t)
#pragma unroll
        for (int r = 0; r < 8; ++r)
            tile[(wave * 16 + 8 * h + r) * 65 + t * 16 + m] = acc[t][r];
    __syncthreads();

    // LN -> exp, in place (one thread per edge-row)
    if (threadIdx.x < 128) {
        const long e = blk0 + threadIdx.x;
        if (e < nedges) {
            float* rp = &tile[threadIdx.x * 65];
            float s = 0.f;
#pragma unroll
            for (int c = 0; c < HH; ++c) s += rp[c];
            const float mu = s * (1.f / HH);
            float vs = 0.f;
#pragma unroll
            for (int c = 0; c < HH; ++c) { const float d = rp[c] - mu; vs += d * d; }
            const float inv = rsqrtf(vs * (1.f / HH) + LN_EPS);
#pragma unroll
            for (int c = 0; c < HH; ++c)
                rp[c] = __expf((rp[c] - mu) * inv * g[c] + b[c]);
        }
    }
    __syncthreads();

    // gather hv[src] * he -> atomic scatter into hsum[dst]
    // 2 threads / edge, 32 contiguous floats each.
    {
        const int el = threadIdx.x >> 1;
        const long e = blk0 + el;
        if (e < nedges) {
            const long s = src[e];
            const long d = dst[e];
            const int  c0 = (threadIdx.x & 1) * 32;
            const float* hvp = hv + s * HH + c0;
            float*       hp  = hsum + d * HH + c0;
            const float* he  = &tile[el * 65 + c0];
#pragma unroll
            for (int c = 0; c < 32; c += 4) {
                const float4 hv4 = *(const float4*)(hvp + c);   // b128 gather
                atomicAdd(&hp[c + 0], hv4.x * he[c + 0]);
                atomicAdd(&hp[c + 1], hv4.y * he[c + 1]);
                atomicAdd(&hp[c + 2], hv4.z * he[c + 2]);
                atomicAdd(&hp[c + 3], hv4.w * he[c + 3]);
            }
        }
    }
}

// =====================================================================
// Kernel 3: out = LN(GELU(h @ W_out))   [nrows x 128]
// 128 thr (4 waves), 64 rows / block.
// =====================================================================
__global__ __launch_bounds__(128) void out_proj_kernel(
    const float* __restrict__ Hin, const v16h* __restrict__ Wp,
    const float* __restrict__ g, const float* __restrict__ b,
    float* __restrict__ out, int nrows)
{
    __shared__ float tile[64 * 129];
    const int wave = threadIdx.x >> 5;
    const int lane = threadIdx.x & 31;
    const int h = lane >> 4, m = lane & 15;
    const int blk0 = blockIdx.x * 64;
    long row = blk0 + wave * 16 + m;
    if (row >= nrows) row = nrows - 1;        // clamp; result discarded

    v8f z = {};
    v8f acc[8];
#pragma unroll
    for (int t = 0; t < 8; ++t) acc[t] = z;

#pragma unroll
    for (int kk = 0; kk < 2; ++kk) {
        const v16h a = load_a_frag(Hin, row, HH, kk * 32, lane);
#pragma unroll
        for (int t = 0; t < 8; ++t) {
            const v16h bf = Wp[(kk * 8 + t) * 32 + lane];   // NT=8
            acc[t] = __builtin_amdgcn_wmma_f32_16x16x32_f16(
                false, a, false, bf, (short)0, acc[t], false, false);
        }
    }

#pragma unroll
    for (int t = 0; t < 8; ++t)
#pragma unroll
        for (int r = 0; r < 8; ++r)
            tile[(wave * 16 + 8 * h + r) * 129 + t * 16 + m] = acc[t][r];
    __syncthreads();

    if (threadIdx.x < 64) {
        const long grow = blk0 + threadIdx.x;
        if (grow < nrows) {
            float* rp = &tile[threadIdx.x * 129];
            float s = 0.f;
#pragma unroll
            for (int c = 0; c < DO; ++c) { float x = gelu_erf(rp[c]); rp[c] = x; s += x; }
            const float mu = s * (1.f / DO);
            float vs = 0.f;
#pragma unroll
            for (int c = 0; c < DO; ++c) { const float d = rp[c] - mu; vs += d * d; }
            const float inv = rsqrtf(vs * (1.f / DO) + LN_EPS);
            float* op = out + grow * DO;
#pragma unroll
            for (int c = 0; c < DO; c += 4) {
                float4 o;
                o.x = (rp[c + 0] - mu) * inv * g[c + 0] + b[c + 0];
                o.y = (rp[c + 1] - mu) * inv * g[c + 1] + b[c + 1];
                o.z = (rp[c + 2] - mu) * inv * g[c + 2] + b[c + 2];
                o.w = (rp[c + 3] - mu) * inv * g[c + 3] + b[c + 3];
                *(float4*)(op + c) = o;
            }
        }
    }
}

// =====================================================================
extern "C" void kernel_launch(void* const* d_in, const int* in_sizes, int n_in,
                              void* d_out, int out_size, void* d_ws, size_t ws_size,
                              hipStream_t stream) {
    const float* node_feats = (const float*)d_in[0];
    const float* edge_feats = (const float*)d_in[1];
    const int*   src        = (const int*)  d_in[2];
    const int*   dst        = (const int*)  d_in[3];
    const float* W_node     = (const float*)d_in[4];
    const float* g_node     = (const float*)d_in[5];
    const float* b_node     = (const float*)d_in[6];
    const float* W_edge     = (const float*)d_in[7];
    const float* g_edge     = (const float*)d_in[8];
    const float* b_edge     = (const float*)d_in[9];
    const float* W_out      = (const float*)d_in[10];
    const float* g_out      = (const float*)d_in[11];
    const float* b_out      = (const float*)d_in[12];

    const int n_nodes = in_sizes[0] / DN;
    const int n_edges = in_sizes[2];

    // workspace layout
    float*    hv    = (float*)d_ws;                          // [n_nodes, 64]
    float*    hsum  = hv + (size_t)n_nodes * HH;             // [n_nodes, 64]
    _Float16* WnP   = (_Float16*)(hsum + (size_t)n_nodes * HH); // 128*64 halves
    _Float16* WeP   = WnP + DN * HH;                         // 128*64 halves
    _Float16* WoP   = WeP + DN * HH;                         // 64*128 halves

    // pack weights -> f16 WMMA B-fragment layout (tiny, L2-resident)
    pack_w_kernel<<<(DN * HH + 255) / 256, 256, 0, stream>>>(W_node, DN, HH, WnP);
    pack_w_kernel<<<(DN * HH + 255) / 256, 256, 0, stream>>>(W_edge, DN, HH, WeP);
    pack_w_kernel<<<(HH * DO + 255) / 256, 256, 0, stream>>>(W_out,  HH, DO, WoP);

    // h accumulator must start at zero every call (atomics add into it)
    hipMemsetAsync(hsum, 0, (size_t)n_nodes * HH * sizeof(float), stream);

    node_proj_kernel<<<(n_nodes + 127) / 128, 256, 0, stream>>>(
        node_feats, (const v16h*)WnP, g_node, b_node, hv, n_nodes);

    edge_fused_kernel<<<(n_edges + 127) / 128, 256, 0, stream>>>(
        edge_feats, (const v16h*)WeP, g_edge, b_edge, src, dst, hv, hsum, n_edges);

    out_proj_kernel<<<(n_nodes + 63) / 64, 128, 0, stream>>>(
        hsum, (const v16h*)WoP, g_out, b_out, (float*)d_out, n_nodes);
}
